// MultiHeadAttention_73306501808525
// MI455X (gfx1250) — matrix operand order repaired and checked by reading, not compiled, not run on hardware
//
#include <hip/hip_runtime.h>
#include <hip/hip_bf16.h>

// ---------------------------------------------------------------------------
// MI455X (gfx1250) multi-head attention forward.
// wave32, WMMA 16x16x32 bf16 (f32 accumulate) for every GEMM.
// Async global->LDS staging (ASYNCcnt) for the P@V operand stream.
// ---------------------------------------------------------------------------

#define BATCH 4
#define SEQ   2048
#define DIN   1024
#define DMODEL 1024
#define HEADS 16
#define DEPTH 64
#define MROWS (BATCH * SEQ)                     // 8192
#define ELEMS_PER_MAT ((size_t)MROWS * DMODEL)  // 8388608
#define ELEMS_PER_W   ((size_t)DMODEL * DIN)    // 1048576

typedef __attribute__((ext_vector_type(16))) __bf16 v16bf;
typedef __attribute__((ext_vector_type(8)))  float  v8f;
typedef __attribute__((ext_vector_type(4)))  int    v4i;
typedef __attribute__((address_space(1))) v4i v4i_g;   // global int4
typedef __attribute__((address_space(3))) v4i v4i_l;   // LDS int4

union BfFrag {
    v16bf v;
    unsigned short u[16];
};

static __device__ __forceinline__ unsigned short f2bf(float f) {
    __bf16 b = (__bf16)f;                      // hardware bf16 convert
    return __builtin_bit_cast(unsigned short, b);
}

// K index inside a 16-bit A-matrix fragment (16x32, ISA 7.12.2):
// lanes 0-15: VGPR0..3 -> K 0..7, VGPR4..7 -> K 16..23; lanes 16-31: +8
static __device__ __forceinline__ int a_kidx(int vv, int hf) {
    int base = (vv < 4) ? (2 * vv) : (16 + 2 * (vv - 4));
    return base + 8 * hf;
}
// K index inside a 16-bit B-matrix fragment (32x16), same order as A:
// lanes 0-15: VGPR0..3 -> K 0..7, VGPR4..7 -> K 16..23; lanes 16-31: +8
static __device__ __forceinline__ int b_kidx(int vv, int hf) {
    int base = (vv < 4) ? (2 * vv) : (16 + 2 * (vv - 4));
    return base + 8 * hf;
}

static __device__ __forceinline__ v8f wmma_bf16(const BfFrag& a, const BfFrag& b, v8f c) {
    return __builtin_amdgcn_wmma_f32_16x16x32_bf16(false, a.v, false, b.v,
                                                   (short)0, c, false, false);
}

// --- CDNA5 async global->LDS copy (16 bytes per lane), tracked by ASYNCcnt
static __device__ __forceinline__ void async_copy_b128(const void* g, void* l) {
#if __has_builtin(__builtin_amdgcn_global_load_async_to_lds_b128)
    __builtin_amdgcn_global_load_async_to_lds_b128(
        (v4i_g*)g, (v4i_l*)l, 0, 0);
#else
    unsigned lofs = (unsigned)(size_t)(v4i_l*)l;
    asm volatile("global_load_async_to_lds_b128 %0, %1, off"
                 :: "v"(lofs), "v"((unsigned long long)g) : "memory");
#endif
}

static __device__ __forceinline__ void wait_asynccnt0() {
#if __has_builtin(__builtin_amdgcn_s_wait_asynccnt)
    __builtin_amdgcn_s_wait_asynccnt(0);
#else
    asm volatile("s_wait_asynccnt 0x0" ::: "memory");
#endif
}

// ---------------------------------------------------------------------------
// Kernel 0: one-time f32 -> bf16 conversion (activations + weights).
// ---------------------------------------------------------------------------
__global__ __launch_bounds__(256) void cvt_bf16_kernel(
    const float* __restrict__ s, unsigned short* __restrict__ d, int n4)
{
    const int i = blockIdx.x * 256 + threadIdx.x;
    if (i >= n4) return;
    const float4 t = ((const float4*)s)[i];
    ushort4 o;
    o.x = f2bf(t.x); o.y = f2bf(t.y); o.z = f2bf(t.z); o.w = f2bf(t.w);
    ((ushort4*)d)[i] = o;
}

// ---------------------------------------------------------------------------
// Kernel 1: fused Q/K/V projection on bf16 operands.  Y = X @ W^T + bias,
// written head-split (B,H,S,DEPTH) bf16.  One wave = 16x64 output tile.
// ---------------------------------------------------------------------------
__global__ __launch_bounds__(256) void qkv_proj_kernel(
    const unsigned short* __restrict__ xq, const unsigned short* __restrict__ xk,
    const unsigned short* __restrict__ xv,
    const unsigned short* __restrict__ wqb, const unsigned short* __restrict__ wkb,
    const unsigned short* __restrict__ wvb,
    const float* __restrict__ bq, const float* __restrict__ bk,
    const float* __restrict__ bv,
    unsigned short* __restrict__ yout)
{
    const int wglobal = blockIdx.x * 8 + (threadIdx.x >> 5);
    const int matrix  = wglobal / 8192;          // 0=Q, 1=K, 2=V
    const int rem     = wglobal % 8192;
    const int mt      = rem & 511;               // 512 row tiles of 16
    const int nt      = rem >> 9;                // 16 col tiles of 64

    const unsigned short* X = (matrix == 0) ? xq : ((matrix == 1) ? xk : xv);
    const unsigned short* W = (matrix == 0) ? wqb : ((matrix == 1) ? wkb : wvb);
    const float* bias       = (matrix == 0) ? bq : ((matrix == 1) ? bk : bv);
    unsigned short* Y       = yout + (size_t)matrix * ELEMS_PER_MAT;

    const int lane = threadIdx.x & 31;
    const int hf   = lane >> 4;
    const int l16  = lane & 15;
    const int mbase = mt * 16;
    const int nbase = nt * 64;

    v8f acc[4] = {v8f{}, v8f{}, v8f{}, v8f{}};

    const int m = mbase + l16;
    for (int kc = 0; kc < DIN; kc += 32) {
        BfFrag a;
#pragma unroll
        for (int vv = 0; vv < 8; ++vv) {
            const int kb = a_kidx(vv, hf);
            a.u[2 * vv]     = X[(size_t)m * DIN + kc + kb];
            a.u[2 * vv + 1] = X[(size_t)m * DIN + kc + kb + 1];
        }
#pragma unroll
        for (int dt = 0; dt < 4; ++dt) {
            const int n = nbase + dt * 16 + l16;
            BfFrag bm;
#pragma unroll
            for (int vv = 0; vv < 8; ++vv) {
                const int kb = b_kidx(vv, hf);
                bm.u[2 * vv]     = W[(size_t)n * DIN + kc + kb];
                bm.u[2 * vv + 1] = W[(size_t)n * DIN + kc + kb + 1];
            }
            acc[dt] = wmma_bf16(a, bm, acc[dt]);
        }
    }

    // Epilogue: + bias, convert to bf16, scatter into (B,H,S,DEPTH).
#pragma unroll
    for (int dt = 0; dt < 4; ++dt) {
        const int n  = nbase + dt * 16 + l16;
        const float bval = bias[n];
        const int h = n >> 6;
        const int d = n & 63;
#pragma unroll
        for (int i = 0; i < 8; ++i) {
            const int r  = mbase + i + 8 * hf;          // C layout: M = i + 8*half
            const int bb = r >> 11;                     // row / SEQ
            const int s  = r & 2047;
            Y[(((size_t)bb * HEADS + h) * SEQ + s) * DEPTH + d] =
                f2bf(acc[dt][i] + bval);
        }
    }
}

// ---------------------------------------------------------------------------
// Kernel 2: attention for one (b,h,16-query-row) block per workgroup.
// 16x2048 f32 logits panel in LDS (128 KB of the WGP's 320 KB), plus a
// double-buffered async-staged V chunk (2 x 8 KB).
// ---------------------------------------------------------------------------
__global__ __launch_bounds__(256) void attn_kernel(
    const unsigned short* __restrict__ qh,
    const unsigned short* __restrict__ kh,
    const unsigned short* __restrict__ vh,
    const float* __restrict__ mask,
    float* __restrict__ attn_out,
    unsigned short* __restrict__ ctx_ws)
{
    __shared__ float ldsF[16 * SEQ];             // logits -> unnormalized exp
    __shared__ float ldsC[16 * DEPTH];           // ctx partial-sum reduction
    __shared__ float ldsInv[16];                 // 1/rowsum
    __shared__ unsigned short ldsV[2][64 * DEPTH]; // double-buffered V chunk

    const int bh    = blockIdx.x >> 7;           // b*H + h
    const int qt    = blockIdx.x & 127;
    const int b     = bh >> 4;
    const int h     = bh & 15;
    const int qbase = qt * 16;

    const int wave = threadIdx.x >> 5;
    const int lane = threadIdx.x & 31;
    const int hf   = lane >> 4;
    const int l16  = lane & 15;

    for (int i = threadIdx.x; i < 16 * DEPTH; i += 256) ldsC[i] = 0.0f;

    // --- Q fragments for this block (16 rows x 64 depth = 2 A-fragments)
    const unsigned short* Qb = qh + ((size_t)bh * SEQ + qbase) * DEPTH;
    BfFrag qf0, qf1;
#pragma unroll
    for (int vv = 0; vv < 8; ++vv) {
        const int kb = a_kidx(vv, hf);
        qf0.u[2 * vv]     = Qb[(size_t)l16 * DEPTH + kb];
        qf0.u[2 * vv + 1] = Qb[(size_t)l16 * DEPTH + kb + 1];
        qf1.u[2 * vv]     = Qb[(size_t)l16 * DEPTH + 32 + kb];
        qf1.u[2 * vv + 1] = Qb[(size_t)l16 * DEPTH + 32 + kb + 1];
    }

    // --- Phase 1: logits = Q K^T / 8 + mask * -1e9 -> LDS
    const unsigned short* Kb = kh + (size_t)bh * SEQ * DEPTH;
    for (int kt = wave; kt < 128; kt += 8) {
        const int n = kt * 16 + l16;             // key position (B column)
        if (kt + 8 < 128)
            __builtin_prefetch(Kb + ((size_t)n + 128) * DEPTH, 0, 3);
        BfFrag b0, b1;
#pragma unroll
        for (int vv = 0; vv < 8; ++vv) {
            const int kb = b_kidx(vv, hf);       // depth index
            b0.u[2 * vv]     = Kb[(size_t)n * DEPTH + kb];
            b0.u[2 * vv + 1] = Kb[(size_t)n * DEPTH + kb + 1];
            b1.u[2 * vv]     = Kb[(size_t)n * DEPTH + 32 + kb];
            b1.u[2 * vv + 1] = Kb[(size_t)n * DEPTH + 32 + kb + 1];
        }
        v8f c = {};
        c = wmma_bf16(qf0, b0, c);
        c = wmma_bf16(qf1, b1, c);
        const float mk = mask[(size_t)b * SEQ + n] * -1.0e9f;
#pragma unroll
        for (int i = 0; i < 8; ++i)
            ldsF[(i + 8 * hf) * SEQ + kt * 16 + l16] = c[i] * 0.125f + mk;
    }
    __syncthreads();

    // --- Phase 2: softmax per row (wave handles 2 rows), stream attn to HBM
    for (int rr = 0; rr < 2; ++rr) {
        const int r = wave * 2 + rr;
        const float4* rowp = (const float4*)(ldsF + (size_t)r * SEQ);
        float4*       roww = (float4*)(ldsF + (size_t)r * SEQ);

        float mx = -3.0e38f;
        for (int j = lane; j < SEQ / 4; j += 32) {
            float4 t = rowp[j];
            mx = fmaxf(mx, fmaxf(fmaxf(t.x, t.y), fmaxf(t.z, t.w)));
        }
#pragma unroll
        for (int o = 16; o > 0; o >>= 1) mx = fmaxf(mx, __shfl_xor(mx, o, 32));

        float sum = 0.0f;
        for (int j = lane; j < SEQ / 4; j += 32) {
            float4 t = rowp[j];
            t.x = __expf(t.x - mx); t.y = __expf(t.y - mx);
            t.z = __expf(t.z - mx); t.w = __expf(t.w - mx);
            sum += (t.x + t.y) + (t.z + t.w);
            roww[j] = t;                         // keep unnormalized exp
        }
#pragma unroll
        for (int o = 16; o > 0; o >>= 1) sum += __shfl_xor(sum, o, 32);
        const float inv = 1.0f / sum;
        if (lane == 0) ldsInv[r] = inv;

        float4* arow = (float4*)(attn_out + (((size_t)bh * SEQ) + qbase + r) * SEQ);
        for (int j = lane; j < SEQ / 4; j += 32) {
            float4 t = rowp[j];
            t.x *= inv; t.y *= inv; t.z *= inv; t.w *= inv;
            arow[j] = t;                         // the 1 GiB mandatory write
        }
    }
    __syncthreads();

    // --- Phase 3: ctx = P @ V.  Each iteration stages a 64-key x 64-depth V
    // chunk into LDS with async b128 copies (double buffered).  Wave w owns
    // (K-half = w>>2, depth-tile = w&3); one WMMA per iteration per wave.
    const unsigned short* Vb = vh + (size_t)bh * SEQ * DEPTH;
    const int myChunk = wave >> 2;               // 0 or 1 (which 32-key half)
    const int dt      = wave & 3;                // depth tile 0..3
    v8f acc = {};

    {   // stage iteration 0 into buffer 0: 256 threads x 2 x 16B = 8 KB
        const int o = threadIdx.x * 8;
        async_copy_b128(Vb + o,        &ldsV[0][o]);
        async_copy_b128(Vb + o + 2048, &ldsV[0][o + 2048]);
    }
    wait_asynccnt0();
    __syncthreads();

    for (int it = 0; it < 32; ++it) {
        const int buf = it & 1;
        if (it + 1 < 32) {                        // prefetch next chunk
            const unsigned short* src = Vb + (size_t)(it + 1) * 64 * DEPTH;
            const int o = threadIdx.x * 8;
            async_copy_b128(src + o,        &ldsV[buf ^ 1][o]);
            async_copy_b128(src + o + 2048, &ldsV[buf ^ 1][o + 2048]);
        }

        const int kg = it * 64 + myChunk * 32;    // global K base for A frag
        BfFrag a;
#pragma unroll
        for (int vv = 0; vv < 8; ++vv) {
            const int kb = a_kidx(vv, hf);
            a.u[2 * vv]     = f2bf(ldsF[(size_t)l16 * SEQ + kg + kb]);
            a.u[2 * vv + 1] = f2bf(ldsF[(size_t)l16 * SEQ + kg + kb + 1]);
        }
        BfFrag bm;
#pragma unroll
        for (int vv = 0; vv < 8; ++vv) {
            const int kb = b_kidx(vv, hf);        // key index within half
            bm.u[2 * vv]     = ldsV[buf][(myChunk * 32 + kb)     * DEPTH + dt * 16 + l16];
            bm.u[2 * vv + 1] = ldsV[buf][(myChunk * 32 + kb + 1) * DEPTH + dt * 16 + l16];
        }
        acc = wmma_bf16(a, bm, acc);

        if (it + 1 < 32) {
            wait_asynccnt0();                     // next buffer landed
            __syncthreads();                      // and everyone done reading
        }
    }

#pragma unroll
    for (int i = 0; i < 8; ++i) {
        const int r = i + 8 * hf;
        atomicAdd(&ldsC[r * DEPTH + dt * 16 + l16], acc[i] * ldsInv[r]);
    }
    __syncthreads();

    // --- Store ctx in concat layout (B,S,H*DEPTH) as bf16
    for (int i = threadIdx.x; i < 16 * DEPTH; i += 256) {
        const int r = i >> 6;
        const int d = i & 63;
        const int s = qbase + r;
        ctx_ws[((size_t)b * SEQ + s) * DMODEL + h * DEPTH + d] = f2bf(ldsC[i]);
    }
}

// ---------------------------------------------------------------------------
// Kernel 3: out = ctx @ dense_w^T + dense_b   (f32 output)
// ---------------------------------------------------------------------------
__global__ __launch_bounds__(256) void dense_kernel(
    const unsigned short* __restrict__ ctx,
    const unsigned short* __restrict__ W,
    const float* __restrict__ bias,
    float* __restrict__ out)
{
    const int wglobal = blockIdx.x * 8 + (threadIdx.x >> 5);
    const int mt = wglobal & 511;
    const int nt = wglobal >> 9;

    const int lane = threadIdx.x & 31;
    const int hf   = lane >> 4;
    const int l16  = lane & 15;
    const int mbase = mt * 16;
    const int nbase = nt * 64;

    v8f acc[4] = {v8f{}, v8f{}, v8f{}, v8f{}};

    const int m = mbase + l16;
    for (int kc = 0; kc < DMODEL; kc += 32) {
        BfFrag a;
#pragma unroll
        for (int vv = 0; vv < 8; ++vv) {
            const int kb = a_kidx(vv, hf);
            a.u[2 * vv]     = ctx[(size_t)m * DMODEL + kc + kb];
            a.u[2 * vv + 1] = ctx[(size_t)m * DMODEL + kc + kb + 1];
        }
#pragma unroll
        for (int dt = 0; dt < 4; ++dt) {
            const int n = nbase + dt * 16 + l16;
            BfFrag bm;
#pragma unroll
            for (int vv = 0; vv < 8; ++vv) {
                const int kb = b_kidx(vv, hf);
                bm.u[2 * vv]     = W[(size_t)n * DMODEL + kc + kb];
                bm.u[2 * vv + 1] = W[(size_t)n * DMODEL + kc + kb + 1];
            }
            acc[dt] = wmma_bf16(a, bm, acc[dt]);
        }
    }

#pragma unroll
    for (int dt = 0; dt < 4; ++dt) {
        const int n = nbase + dt * 16 + l16;
        const float bval = bias[n];
#pragma unroll
        for (int i = 0; i < 8; ++i) {
            const int r = mbase + i + 8 * hf;
            out[(size_t)r * DMODEL + n] = acc[dt][i] + bval;
        }
    }
}

// ---------------------------------------------------------------------------
extern "C" void kernel_launch(void* const* d_in, const int* in_sizes, int n_in,
                              void* d_out, int out_size, void* d_ws, size_t ws_size,
                              hipStream_t stream) {
    const float* vin  = (const float*)d_in[0];
    const float* kin  = (const float*)d_in[1];
    const float* qin  = (const float*)d_in[2];
    const float* mask = (const float*)d_in[3];
    const float* wq   = (const float*)d_in[4];
    const float* bq   = (const float*)d_in[5];
    const float* wk   = (const float*)d_in[6];
    const float* bk   = (const float*)d_in[7];
    const float* wv   = (const float*)d_in[8];
    const float* bv   = (const float*)d_in[9];
    const float* wd   = (const float*)d_in[10];
    const float* bd   = (const float*)d_in[11];

    // Workspace layout (bf16 / ushort elements), ~126 MB total:
    unsigned short* ws  = (unsigned short*)d_ws;
    unsigned short* xq  = ws;                              // converted inputs
    unsigned short* xk  = xq  + ELEMS_PER_MAT;
    unsigned short* xv  = xk  + ELEMS_PER_MAT;
    unsigned short* wqb = xv  + ELEMS_PER_MAT;             // converted weights
    unsigned short* wkb = wqb + ELEMS_PER_W;
    unsigned short* wvb = wkb + ELEMS_PER_W;
    unsigned short* wdb = wvb + ELEMS_PER_W;
    unsigned short* qh  = wdb + ELEMS_PER_W;               // head-split Q/K/V
    unsigned short* kh  = qh  + ELEMS_PER_MAT;
    unsigned short* vh  = kh  + ELEMS_PER_MAT;
    unsigned short* ctx = vh  + ELEMS_PER_MAT;

    float* out  = (float*)d_out;                           // (B,S,DMODEL)
    float* attn = out + ELEMS_PER_MAT;                     // (B,H,S,S)

    const int n4x = (int)(ELEMS_PER_MAT / 4);              // 2097152
    const int n4w = (int)(ELEMS_PER_W / 4);                // 262144
    cvt_bf16_kernel<<<dim3(n4x / 256), dim3(256), 0, stream>>>(qin, xq, n4x);
    cvt_bf16_kernel<<<dim3(n4x / 256), dim3(256), 0, stream>>>(kin, xk, n4x);
    cvt_bf16_kernel<<<dim3(n4x / 256), dim3(256), 0, stream>>>(vin, xv, n4x);
    cvt_bf16_kernel<<<dim3(n4w / 256), dim3(256), 0, stream>>>(wq, wqb, n4w);
    cvt_bf16_kernel<<<dim3(n4w / 256), dim3(256), 0, stream>>>(wk, wkb, n4w);
    cvt_bf16_kernel<<<dim3(n4w / 256), dim3(256), 0, stream>>>(wv, wvb, n4w);
    cvt_bf16_kernel<<<dim3(n4w / 256), dim3(256), 0, stream>>>(wd, wdb, n4w);

    qkv_proj_kernel<<<dim3(3072), dim3(256), 0, stream>>>(
        xq, xk, xv, wqb, wkb, wvb, bq, bk, bv, qh);
    attn_kernel<<<dim3(BATCH * HEADS * (SEQ / 16)), dim3(256), 0, stream>>>(
        qh, kh, vh, mask, attn, ctx);
    dense_kernel<<<dim3(1024), dim3(256), 0, stream>>>(ctx, wdb, bd, out);
}
